// RelativePositionMultiHeadAttention_7026566496473
// MI455X (gfx1250) — matrix operand order repaired and checked by reading, not compiled, not run on hardware
//
#include <hip/hip_runtime.h>

typedef __attribute__((ext_vector_type(16))) __bf16 v16bf;
typedef __attribute__((ext_vector_type(8)))  __bf16 v8bf;
typedef __attribute__((ext_vector_type(8)))  float  v8f;
typedef __attribute__((ext_vector_type(4)))  float  v4f;

#define D_MODEL 1024
#define NUM_HEADS 16
#define D_K 64
#define MAX_LEN 2048
#define BATCH 2
#define SEQ 1024

static __device__ __forceinline__ __bf16 f2bf(float f) {
    unsigned u = __builtin_bit_cast(unsigned, f);
    unsigned r = u + 0x7FFFu + ((u >> 16) & 1u);
    unsigned short h = (unsigned short)(r >> 16);
    return __builtin_bit_cast(__bf16, h);
}
static __device__ __forceinline__ float bf2f(__bf16 b) {
    unsigned short h = __builtin_bit_cast(unsigned short, b);
    unsigned u = ((unsigned)h) << 16;
    return __builtin_bit_cast(float, u);
}

// ---------------------------------------------------------------- fp32 -> bf16
// Vectorized: 8 elements / thread-iteration (2x b128 load, 1x b128 store)
__global__ void cvt_bf16_kernel(const float* __restrict__ in,
                                __bf16* __restrict__ out, int n8) {
    int i = blockIdx.x * blockDim.x + threadIdx.x;
    int stride = gridDim.x * blockDim.x;
    for (; i < n8; i += stride) {
        v4f lo = *(const v4f*)(in + (size_t)i * 8);
        v4f hi = *(const v4f*)(in + (size_t)i * 8 + 4);
        v8bf o;
#pragma unroll
        for (int k = 0; k < 4; ++k) { o[k] = f2bf(lo[k]); o[4 + k] = f2bf(hi[k]); }
        *(v8bf*)(out + (size_t)i * 8) = o;
    }
}

// Build the WMMA A-operand (16x32 bf16) for this lane from a row pointer.
// Lanes 0-15: row l16, K {k0..k0+15}; lanes 16-31: K {k0+16..k0+31}.
static __device__ __forceinline__ v16bf loadA(const __bf16* rowBase, int k0, int laneHi) {
    int ab = k0 + laneHi * 16;
    v8bf lo = *(const v8bf*)(rowBase + ab);
    v8bf hi = *(const v8bf*)(rowBase + ab + 8);
    v16bf a;
#pragma unroll
    for (int i = 0; i < 8; ++i) { a[i] = lo[i]; a[8 + i] = hi[i]; }
    return a;
}

// ---------------------------------------------------------------- projections
// 32x64 output tile per wave: 2 M-subtiles x 4 N-subtiles, 8 accumulators.
// y[row, e] = sum_d xb[row, d] * W[e, d]  (NT GEMM)
// mode 0/1: store bf16 [b,h,s,dk] ; mode 2: store bf16 transposed [b,h,dk,s]
__global__ void __launch_bounds__(128)
proj_kernel(const __bf16* __restrict__ xb, const __bf16* __restrict__ w,
            const float* __restrict__ bias, __bf16* __restrict__ out,
            int mode, float scale) {
    const int lane   = threadIdx.x & 31;
    const int laneHi = lane >> 4;
    const int l16    = lane & 15;
    const int waveId = blockIdx.x * (blockDim.x >> 5) + (threadIdx.x >> 5);
    const int tn = waveId & 15;           // 16 tiles of 64 cols
    const int tm = waveId >> 4;           // 64 tiles of 32 rows over B*S
    const int row0 = tm * 32;
    const int e0   = tn * 64;

    const __bf16* aRow0 = xb + (size_t)(row0 + l16) * D_MODEL;
    const __bf16* aRow1 = aRow0 + (size_t)16 * D_MODEL;
    const __bf16* bRow[4];
#pragma unroll
    for (int ns = 0; ns < 4; ++ns)
        bRow[ns] = w + (size_t)(e0 + ns * 16 + l16) * D_MODEL + laneHi * 16;

    v8f acc[2][4] = {};
    for (int k0 = 0; k0 < D_MODEL; k0 += 32) {
        if (k0 + 256 < D_MODEL) {
            __builtin_prefetch(aRow0 + k0 + 256, 0, 3);
            __builtin_prefetch(aRow1 + k0 + 256, 0, 3);
#pragma unroll
            for (int ns = 0; ns < 4; ++ns) __builtin_prefetch(bRow[ns] + k0 + 256, 0, 3);
        }
        v16bf a0 = loadA(aRow0, k0, laneHi);
        v16bf a1 = loadA(aRow1, k0, laneHi);
#pragma unroll
        for (int ns = 0; ns < 4; ++ns) {
            v16bf b = *(const v16bf*)(bRow[ns] + k0);
            acc[0][ns] = __builtin_amdgcn_wmma_f32_16x16x32_bf16(false, a0, false, b,
                             (short)0, acc[0][ns], false, false);
            acc[1][ns] = __builtin_amdgcn_wmma_f32_16x16x32_bf16(false, a1, false, b,
                             (short)0, acc[1][ns], false, false);
        }
    }

    const int bidx  = row0 / SEQ;
    const int srow  = row0 % SEQ;
    const int h     = tn;                 // e0 = tn*64, dk = 64
    const int mBase = laneHi * 8;
#pragma unroll
    for (int ns = 0; ns < 4; ++ns) {
        const float bval = bias[e0 + ns * 16 + l16];
        const int ed = ns * 16 + l16;
#pragma unroll
        for (int ms = 0; ms < 2; ++ms) {
            if (mode == 2) {  // V^T: contiguous along s
                __bf16* p = out + ((size_t)((bidx * NUM_HEADS + h) * D_K + ed)) * SEQ
                                + srow + ms * 16 + mBase;
                v8bf vv;
#pragma unroll
                for (int r = 0; r < 8; ++r) vv[r] = f2bf((acc[ms][ns][r] + bval) * scale);
                *(v8bf*)p = vv;
            } else {          // Q / K: [b,h,s,dk]
#pragma unroll
                for (int r = 0; r < 8; ++r) {
                    int s = srow + ms * 16 + mBase + r;
                    out[((size_t)((bidx * NUM_HEADS + h) * SEQ + s)) * D_K + ed] =
                        f2bf((acc[ms][ns][r] + bval) * scale);
                }
            }
        }
    }
}

// ---------------------------------------------------------------- attention
__global__ void __launch_bounds__(256)
attn_kernel(const __bf16* __restrict__ q, const __bf16* __restrict__ kmat,
            const __bf16* __restrict__ vT, const __bf16* __restrict__ er,
            __bf16* __restrict__ ao) {
    __shared__ __bf16 sc[16][SEQ];        // 32 KB: scores -> exp(scores)
    __shared__ float  gs[8][16][32];      // 16 KB: BD scratch, then PV partials
    __shared__ float  redmax[16][16];
    __shared__ float  redsum[16][16];
    __shared__ float  rowinv[16];

    const int it = blockIdx.x;
    const int h  = blockIdx.y;
    const int b  = blockIdx.z;
    const int bh = b * NUM_HEADS + h;
    const int i0 = it * 16;

    const int tid    = threadIdx.x;
    const int lane   = tid & 31;
    const int laneHi = lane >> 4;
    const int l16    = lane & 15;
    const int w      = tid >> 5;
    const int mBase  = laneHi * 8;

    const __bf16* qRow = q + ((size_t)bh * SEQ + i0 + l16) * D_K;
    v16bf aq0 = loadA(qRow, 0, laneHi);
    v16bf aq1 = loadA(qRow, 32, laneHi);

    // ---- Phase 1: scores = Q·K^T + Q·Er[S-1+i-j]  (1/sqrt(dk) folded into Q)
    for (int jt = w; jt < SEQ / 16; jt += 8) {
        const int j0 = jt * 16;

        const __bf16* kRow = kmat + ((size_t)bh * SEQ + j0 + l16) * D_K + laneHi * 16;
        v8f ac = {};
        ac = __builtin_amdgcn_wmma_f32_16x16x32_bf16(false, aq0, false,
                 *(const v16bf*)(kRow + 0), (short)0, ac, false, false);
        ac = __builtin_amdgcn_wmma_f32_16x16x32_bf16(false, aq1, false,
                 *(const v16bf*)(kRow + 32), (short)0, ac, false, false);

        // G[di,p] = Q[i0+di] · Er[rbase + p],  BD[di,dj] = G[di, 15+di-dj]
        const int rbase = (SEQ - 1) + i0 - j0 - 15;   // always in bounds
#pragma unroll
        for (int ps = 0; ps < 2; ++ps) {
            const __bf16* eRow = er + (size_t)(rbase + ps * 16 + l16) * D_K + laneHi * 16;
            v8f g = {};
            g = __builtin_amdgcn_wmma_f32_16x16x32_bf16(false, aq0, false,
                    *(const v16bf*)(eRow + 0), (short)0, g, false, false);
            g = __builtin_amdgcn_wmma_f32_16x16x32_bf16(false, aq1, false,
                    *(const v16bf*)(eRow + 32), (short)0, g, false, false);
#pragma unroll
            for (int r = 0; r < 8; ++r) gs[w][mBase + r][ps * 16 + l16] = g[r];
        }
#pragma unroll
        for (int r = 0; r < 8; ++r) {
            int m = mBase + r;
            float bd = gs[w][m][15 + m - l16];
            sc[m][j0 + l16] = f2bf(ac[r] + bd);
        }
    }
    __syncthreads();

    // ---- Phase 2: softmax per row (16 threads per row)
    const int rr  = tid >> 4;
    const int c16 = tid & 15;
    float mx = -3.0e38f;
    for (int j = c16; j < SEQ; j += 16) mx = fmaxf(mx, bf2f(sc[rr][j]));
    redmax[rr][c16] = mx;
    __syncthreads();
    float rmax = redmax[rr][0];
#pragma unroll
    for (int i = 1; i < 16; ++i) rmax = fmaxf(rmax, redmax[rr][i]);
    float sum = 0.f;
    for (int j = c16; j < SEQ; j += 16) {
        float e = __expf(bf2f(sc[rr][j]) - rmax);
        sum += e;
        sc[rr][j] = f2bf(e);
    }
    redsum[rr][c16] = sum;
    __syncthreads();
    if (tid < 16) {
        float s = 0.f;
#pragma unroll
        for (int i = 0; i < 16; ++i) s += redsum[tid][i];
        rowinv[tid] = 1.0f / s;
    }
    __syncthreads();

    // ---- Phase 3: out = P · V  -- all 8 waves: w&3 -> dk subtile, w>>2 -> j half
    {
        const int ns    = w & 3;
        const int jbeg  = (w >> 2) * (SEQ / 2);
        const int e0    = ns * 16;
        const __bf16* vRow = vT + ((size_t)bh * D_K + e0 + l16) * SEQ + laneHi * 16;
        v8f acc = {};
        for (int j0 = jbeg; j0 < jbeg + SEQ / 2; j0 += 32) {
            v16bf a = loadA(&sc[l16][0], j0, laneHi);
            v16bf bB = *(const v16bf*)(vRow + j0);
            acc = __builtin_amdgcn_wmma_f32_16x16x32_bf16(false, a, false, bB,
                                                          (short)0, acc, false, false);
        }
        float* gsf = &gs[0][0][0];        // reuse BD scratch for PV partials
        if (w >= 4) {
#pragma unroll
            for (int r = 0; r < 8; ++r)
                gsf[(((w - 4) * 16) + mBase + r) * 16 + l16] = acc[r];
        }
        __syncthreads();
        if (w < 4) {
#pragma unroll
            for (int r = 0; r < 8; ++r) {
                int m = mBase + r;
                float val = (acc[r] + gsf[((w * 16) + m) * 16 + l16]) * rowinv[m];
                ao[((size_t)(b * SEQ + i0 + m)) * D_MODEL + h * D_K + e0 + l16] = f2bf(val);
            }
        }
    }
}

// ---------------------------------------------------------------- out proj
// Same 32x64-per-wave tiling, fp32 output with bias.
__global__ void __launch_bounds__(128)
oproj_kernel(const __bf16* __restrict__ ab, const __bf16* __restrict__ wo,
             const float* __restrict__ bo, float* __restrict__ out) {
    const int lane   = threadIdx.x & 31;
    const int laneHi = lane >> 4;
    const int l16    = lane & 15;
    const int waveId = blockIdx.x * (blockDim.x >> 5) + (threadIdx.x >> 5);
    const int tn = waveId & 15;
    const int tm = waveId >> 4;
    const int row0 = tm * 32;
    const int e0   = tn * 64;

    const __bf16* aRow0 = ab + (size_t)(row0 + l16) * D_MODEL;
    const __bf16* aRow1 = aRow0 + (size_t)16 * D_MODEL;
    const __bf16* bRow[4];
#pragma unroll
    for (int ns = 0; ns < 4; ++ns)
        bRow[ns] = wo + (size_t)(e0 + ns * 16 + l16) * D_MODEL + laneHi * 16;

    v8f acc[2][4] = {};
    for (int k0 = 0; k0 < D_MODEL; k0 += 32) {
        if (k0 + 256 < D_MODEL) {
            __builtin_prefetch(aRow0 + k0 + 256, 0, 3);
            __builtin_prefetch(aRow1 + k0 + 256, 0, 3);
#pragma unroll
            for (int ns = 0; ns < 4; ++ns) __builtin_prefetch(bRow[ns] + k0 + 256, 0, 3);
        }
        v16bf a0 = loadA(aRow0, k0, laneHi);
        v16bf a1 = loadA(aRow1, k0, laneHi);
#pragma unroll
        for (int ns = 0; ns < 4; ++ns) {
            v16bf b = *(const v16bf*)(bRow[ns] + k0);
            acc[0][ns] = __builtin_amdgcn_wmma_f32_16x16x32_bf16(false, a0, false, b,
                             (short)0, acc[0][ns], false, false);
            acc[1][ns] = __builtin_amdgcn_wmma_f32_16x16x32_bf16(false, a1, false, b,
                             (short)0, acc[1][ns], false, false);
        }
    }
    const int mBase = laneHi * 8;
#pragma unroll
    for (int ns = 0; ns < 4; ++ns) {
        const float bval = bo[e0 + ns * 16 + l16];
#pragma unroll
        for (int ms = 0; ms < 2; ++ms)
#pragma unroll
            for (int r = 0; r < 8; ++r)
                out[(size_t)(row0 + ms * 16 + mBase + r) * D_MODEL + e0 + ns * 16 + l16] =
                    acc[ms][ns][r] + bval;
    }
}

// ---------------------------------------------------------------- launcher
extern "C" void kernel_launch(void* const* d_in, const int* in_sizes, int n_in,
                              void* d_out, int out_size, void* d_ws, size_t ws_size,
                              hipStream_t stream) {
    const float* x  = (const float*)d_in[0];
    const float* Wq = (const float*)d_in[1];
    const float* bq = (const float*)d_in[2];
    const float* Wk = (const float*)d_in[3];
    const float* bk = (const float*)d_in[4];
    const float* Wv = (const float*)d_in[5];
    const float* bv = (const float*)d_in[6];
    const float* Wo = (const float*)d_in[7];
    const float* bo = (const float*)d_in[8];
    const float* Er = (const float*)d_in[9];
    float* out = (float*)d_out;

    const int NX = BATCH * SEQ * D_MODEL;
    const int NW = D_MODEL * D_MODEL;
    const int NE = MAX_LEN * D_K;
    const int NQ = BATCH * NUM_HEADS * SEQ * D_K;

    char* ws = (char*)d_ws;
    size_t off = 0;
    auto alloc = [&](size_t elems) {
        char* p = ws + off;
        off = (off + elems * sizeof(__bf16) + 255) & ~(size_t)255;
        return (__bf16*)p;
    };
    __bf16* xb  = alloc(NX);
    __bf16* wqb = alloc(NW);
    __bf16* wkb = alloc(NW);
    __bf16* wvb = alloc(NW);
    __bf16* wob = alloc(NW);
    __bf16* erb = alloc(NE);
    __bf16* qb  = alloc(NQ);
    __bf16* kb  = alloc(NQ);
    __bf16* vtb = alloc(NQ);
    __bf16* aob = alloc(NX);

    cvt_bf16_kernel<<<1024, 256, 0, stream>>>(x,  xb,  NX / 8);
    cvt_bf16_kernel<<<512,  256, 0, stream>>>(Wq, wqb, NW / 8);
    cvt_bf16_kernel<<<512,  256, 0, stream>>>(Wk, wkb, NW / 8);
    cvt_bf16_kernel<<<512,  256, 0, stream>>>(Wv, wvb, NW / 8);
    cvt_bf16_kernel<<<512,  256, 0, stream>>>(Wo, wob, NW / 8);
    cvt_bf16_kernel<<<64,   256, 0, stream>>>(Er, erb, NE / 8);

    // 64 x 16 = 1024 waves of 32x64 tiles, 4 waves per block
    proj_kernel<<<256, 128, 0, stream>>>(xb, wqb, bq, qb,  0, 0.125f); // Q / sqrt(dk)
    proj_kernel<<<256, 128, 0, stream>>>(xb, wkb, bk, kb,  1, 1.0f);   // K
    proj_kernel<<<256, 128, 0, stream>>>(xb, wvb, bv, vtb, 2, 1.0f);   // V^T

    attn_kernel<<<dim3(SEQ / 16, NUM_HEADS, BATCH), 256, 0, stream>>>(
        qb, kb, vtb, erb, aob);

    oproj_kernel<<<256, 128, 0, stream>>>(aob, wob, bo, out);
}
